// BaselineMona_26139170963647
// MI455X (gfx1250) — matrix operand order repaired and checked by reading, not compiled, not run on hardware
//
#include <hip/hip_runtime.h>
#include <hip/hip_bf16.h>
#include <math.h>

// ---------------------------------------------------------------------------
// Mona adapter block for MI455X (gfx1250, wave32, WMMA).
// B=64, N=1025, D=1024, C=64, H=W=32.
// GEMMs + 1x1 proj run on v_wmma_f32_16x16x32_bf16 (f32 accumulate).
// ---------------------------------------------------------------------------

typedef __bf16 bf16_t;
typedef __attribute__((ext_vector_type(16))) __bf16 v16bf;
typedef __attribute__((ext_vector_type(8)))  float  v8f;

#define BDIM  64
#define NTOK  1025
#define DDIM  1024
#define CDIM  64
#define HW    32
#define NTOTAL (BDIM * NTOK)      // 65600 tokens
#define LDA1  1048                // padded LDS stride for k1 A tile (bank-conflict free)
#define LDA3  80                  // padded LDS stride for k3 A tile

static __device__ __forceinline__ bf16_t f2bf(float f) { return (bf16_t)f; }

// ---------------------------------------------------------------------------
// k0: weight prep.  w1t[c][d] = bf16(w1[d][c]);  w2t[d][c] = bf16(w2[c][d]).
// Makes WMMA B-fragments contiguous-in-K (two 16B loads per lane).
// ---------------------------------------------------------------------------
__global__ __launch_bounds__(256) void k0_prep(const float* __restrict__ w1,
                                               const float* __restrict__ w2,
                                               bf16_t* __restrict__ w1t,
                                               bf16_t* __restrict__ w2t) {
  int i = blockIdx.x * 256 + threadIdx.x;           // 0..65535
  int c1 = i >> 10, d1 = i & 1023;
  w1t[i] = f2bf(w1[d1 * CDIM + c1]);
  int d2 = i >> 6, c2 = i & 63;
  w2t[i] = f2bf(w2[c2 * DDIM + d2]);
}

// ---------------------------------------------------------------------------
// k1: LayerNorm + (ln*gamma + x*gammax) + GEMM1 (D=1024 -> C=64) + b1.
// 128 threads = 4 waves; 16 tokens per block; each wave owns one 16-col tile.
// ---------------------------------------------------------------------------
__global__ __launch_bounds__(128) void k1_ln_gemm(
    const float* __restrict__ x,     const float* __restrict__ ln_w,
    const float* __restrict__ ln_b,  const float* __restrict__ gamma,
    const float* __restrict__ gammax,
    const bf16_t* __restrict__ w1t,  const float* __restrict__ b1,
    float* __restrict__ y1) {
  __shared__ bf16_t As[16 * LDA1];
  __shared__ float  redS[16][8], redQ[16][8];
  __shared__ float  smu[16], srs[16];

  const int tid = threadIdx.x;
  const int row = tid >> 3, sub = tid & 7;          // 8 threads per token row
  const int tokenBase = blockIdx.x * 16;
  const int token = tokenBase + row;
  const int bb = token / NTOK, nn = token % NTOK;
  const float* xr = x + (size_t)nn * (BDIM * DDIM) + (size_t)bb * DDIM;

  // ---- pass 1: per-token mean / var over D=1024 ----
  float s = 0.f, q = 0.f;
  const float4* x4 = (const float4*)(xr + sub * 128);
#pragma unroll 8
  for (int i = 0; i < 32; ++i) {
    float4 v = x4[i];
    s += v.x + v.y + v.z + v.w;
    q += v.x * v.x + v.y * v.y + v.z * v.z + v.w * v.w;
  }
  redS[row][sub] = s; redQ[row][sub] = q;
  __syncthreads();
  if (sub == 0) {
    float ts = 0.f, tq = 0.f;
#pragma unroll
    for (int i = 0; i < 8; ++i) { ts += redS[row][i]; tq += redQ[row][i]; }
    float mu = ts * (1.f / 1024.f);
    float var = tq * (1.f / 1024.f) - mu * mu;
    smu[row] = mu;
    srs[row] = rsqrtf(var + 1e-5f);
  }
  __syncthreads();

  // ---- pass 2: normalize + scale -> bf16 A tile in LDS ----
  const float mu = smu[row], rs = srs[row];
  for (int i = 0; i < 128; i += 4) {
    int d = sub * 128 + i;
    float4 v  = *(const float4*)(xr + d);
    float4 lw = *(const float4*)(ln_w + d);
    float4 lb = *(const float4*)(ln_b + d);
    float4 gm = *(const float4*)(gamma + d);
    float4 gx = *(const float4*)(gammax + d);
    float a0 = (v.x - mu) * rs * lw.x + lb.x;
    float a1 = (v.y - mu) * rs * lw.y + lb.y;
    float a2 = (v.z - mu) * rs * lw.z + lb.z;
    float a3 = (v.w - mu) * rs * lw.w + lb.w;
    bf16_t* ap = As + row * LDA1 + d;
    ap[0] = f2bf(a0 * gm.x + v.x * gx.x);
    ap[1] = f2bf(a1 * gm.y + v.y * gx.y);
    ap[2] = f2bf(a2 * gm.z + v.z * gx.z);
    ap[3] = f2bf(a3 * gm.w + v.w * gx.w);
  }
  __syncthreads();

  // ---- GEMM: 16x64 = (16x1024) x (1024x64), 32 K-steps of 16x16x32 ----
  const int lane = tid & 31;
  const int wv   = tid >> 5;                 // col tile 0..3
  const int nB   = wv * 16;
  const int m    = lane & 15;
  const int kh   = lane >> 4;
  const int hi   = (lane & 16) ? 8 : 0;

  v8f acc = {};
#pragma unroll 4
  for (int kk = 0; kk < 32; ++kk) {
    const int k0 = kk * 32;
    v16bf a, bfr;
    const bf16_t* ap = As + m * LDA1 + k0 + kh * 8;
#pragma unroll
    for (int j = 0; j < 8; ++j) { a[j] = ap[j]; a[8 + j] = ap[16 + j]; }
    const bf16_t* bp = w1t + (size_t)(nB + m) * DDIM + k0 + kh * 8;
#pragma unroll
    for (int j = 0; j < 8; ++j) { bfr[j] = bp[j]; bfr[8 + j] = bp[16 + j]; }
    acc = __builtin_amdgcn_wmma_f32_16x16x32_bf16(false, a, false, bfr,
                                                  (short)0, acc, false, false);
  }

  const int col = nB + m;
  const float bc = b1[col];
#pragma unroll
  for (int v = 0; v < 8; ++v) {
    int t2 = tokenBase + v + hi;
    y1[(size_t)t2 * CDIM + col] = acc[v] + bc;
  }
}

// ---------------------------------------------------------------------------
// k2: depthwise 3x3 + 5x5 + 7x7 (SAME, zero pad), averaged, + identity.
// NHWC layout, float4 over channels.  mix written as f32 and bf16.
// ---------------------------------------------------------------------------
__global__ __launch_bounds__(256) void k2_conv(
    const float* __restrict__ y1,
    const float* __restrict__ w3, const float* __restrict__ b3,
    const float* __restrict__ w5, const float* __restrict__ b5,
    const float* __restrict__ w7, const float* __restrict__ b7,
    float* __restrict__ mixf, bf16_t* __restrict__ mixb) {
  __shared__ float w3s[CDIM * 9], w5s[CDIM * 25], w7s[CDIM * 49];
  const int tid = threadIdx.x;
  for (int i = tid; i < CDIM * 9;  i += 256) w3s[i] = w3[i];
  for (int i = tid; i < CDIM * 25; i += 256) w5s[i] = w5[i];
  for (int i = tid; i < CDIM * 49; i += 256) w7s[i] = w7[i];
  __syncthreads();

  const int gp = blockIdx.x;                  // 4096 blocks
  const int b  = gp >> 6;
  const int pixBase = (gp & 63) * 16;
  const int pl = tid >> 4;                    // pixel in tile
  const int cg = (tid & 15) * 4;              // channel group
  const int p  = pixBase + pl;
  const int r  = p >> 5, cl = p & 31;
  const float* img = y1 + ((size_t)b * NTOK + 1) * CDIM;   // spatial start

  float a3[4] = {0,0,0,0}, a5[4] = {0,0,0,0}, a7[4] = {0,0,0,0};
  float ctr[4] = {0,0,0,0};
  for (int dy = -3; dy <= 3; ++dy) {
    int rr = r + dy; if (rr < 0 || rr >= HW) continue;
    for (int dx = -3; dx <= 3; ++dx) {
      int cc = cl + dx; if (cc < 0 || cc >= HW) continue;
      float4 v = *(const float4*)(img + (size_t)(rr * HW + cc) * CDIM + cg);
      float vv[4] = {v.x, v.y, v.z, v.w};
      const int i7 = (dy + 3) * 7 + (dx + 3);
      const bool in5 = (dy >= -2 && dy <= 2 && dx >= -2 && dx <= 2);
      const int i5 = (dy + 2) * 5 + (dx + 2);
      const bool in3 = (dy >= -1 && dy <= 1 && dx >= -1 && dx <= 1);
      const int i3 = (dy + 1) * 3 + (dx + 1);
#pragma unroll
      for (int j = 0; j < 4; ++j) {
        int c = cg + j;
        a7[j] += vv[j] * w7s[c * 49 + i7];
        if (in5) a5[j] += vv[j] * w5s[c * 25 + i5];
        if (in3) a3[j] += vv[j] * w3s[c * 9 + i3];
      }
      if (dy == 0 && dx == 0) { ctr[0]=vv[0]; ctr[1]=vv[1]; ctr[2]=vv[2]; ctr[3]=vv[3]; }
    }
  }
  const size_t o = ((size_t)b * 1024 + p) * CDIM + cg;
#pragma unroll
  for (int j = 0; j < 4; ++j) {
    int c = cg + j;
    float mv = (a3[j] + b3[c] + a5[j] + b5[c] + a7[j] + b7[c]) * (1.f / 3.f) + ctr[j];
    mixf[o + j] = mv;
    mixb[o + j] = f2bf(mv);
  }
}

// ---------------------------------------------------------------------------
// k2b: 1x1 proj as WMMA GEMM: mix(65536,64) x proj_w^T(64,64), residual
// written in place into mixf.  128 threads = 4 waves x 16-col tiles.
// ---------------------------------------------------------------------------
__global__ __launch_bounds__(128) void k2b_proj(
    const bf16_t* __restrict__ mixb, const float* __restrict__ projw,
    const float* __restrict__ projb, float* __restrict__ mixf) {
  const int tid = threadIdx.x, lane = tid & 31, wv = tid >> 5;
  const int pixBase = blockIdx.x * 16;        // 4096 blocks
  const int nB = wv * 16;
  const int m = lane & 15, kh = lane >> 4;
  const int hi = (lane & 16) ? 8 : 0;

  v8f acc = {};
#pragma unroll
  for (int kc = 0; kc < 2; ++kc) {
    const int k0 = kc * 32;
    v16bf a, bfr;
    const bf16_t* ap = mixb + (size_t)(pixBase + m) * CDIM + k0 + kh * 8;
#pragma unroll
    for (int j = 0; j < 8; ++j) { a[j] = ap[j]; a[8 + j] = ap[16 + j]; }
    const float* bp = projw + (size_t)(nB + m) * CDIM + k0 + kh * 8;  // proj_w[o][c]
#pragma unroll
    for (int j = 0; j < 8; ++j) { bfr[j] = f2bf(bp[j]); bfr[8 + j] = f2bf(bp[16 + j]); }
    acc = __builtin_amdgcn_wmma_f32_16x16x32_bf16(false, a, false, bfr,
                                                  (short)0, acc, false, false);
  }
  const int col = nB + m;
  const float pb = projb[col];
#pragma unroll
  for (int v = 0; v < 8; ++v) {
    size_t o = (size_t)(pixBase + v + hi) * CDIM + col;
    mixf[o] = mixf[o] + acc[v] + pb;          // sp = sp + proj
  }
}

// ---------------------------------------------------------------------------
// k3: exact GELU + GEMM2 (C=64 -> D=1024) + b2 + residual(x), output (N,B,D).
// 256 threads = 8 waves; 16 tokens per block; 8 passes over D.
// ---------------------------------------------------------------------------
__global__ __launch_bounds__(256) void k3_gelu_gemm(
    const float* __restrict__ y1, const float* __restrict__ sp,
    const bf16_t* __restrict__ w2t, const float* __restrict__ b2,
    const float* __restrict__ x, float* __restrict__ out) {
  __shared__ bf16_t As[16 * LDA3];
  const int tid = threadIdx.x;
  const int tokenBase = blockIdx.x * 16;

  // GELU into bf16 A tile (cls from y1, spatial from sp)
  for (int i = tid * 4; i < 16 * CDIM; i += 256 * 4) {
#pragma unroll
    for (int j = 0; j < 4; ++j) {
      int idx = i + j;
      int row = idx >> 6, c = idx & 63;
      int token = tokenBase + row;
      int b = token / NTOK, n = token % NTOK;
      float v = (n == 0) ? y1[(size_t)token * CDIM + c]
                         : sp[((size_t)b * 1024 + (n - 1)) * CDIM + c];
      float g = 0.5f * v * (1.f + erff(v * 0.70710678118654752f));
      As[row * LDA3 + c] = f2bf(g);
    }
  }
  __syncthreads();

  const int lane = tid & 31, wv = tid >> 5;
  const int m = lane & 15, kh = lane >> 4;
  const int hi = (lane & 16) ? 8 : 0;

  for (int pass = 0; pass < 8; ++pass) {
    const int dB = pass * 128 + wv * 16;
    v8f acc = {};
#pragma unroll
    for (int kc = 0; kc < 2; ++kc) {
      const int k0 = kc * 32;
      v16bf a, bfr;
      const bf16_t* ap = As + m * LDA3 + k0 + kh * 8;
#pragma unroll
      for (int j = 0; j < 8; ++j) { a[j] = ap[j]; a[8 + j] = ap[16 + j]; }
      const bf16_t* bp = w2t + (size_t)(dB + m) * CDIM + k0 + kh * 8;
#pragma unroll
      for (int j = 0; j < 8; ++j) { bfr[j] = bp[j]; bfr[8 + j] = bp[16 + j]; }
      acc = __builtin_amdgcn_wmma_f32_16x16x32_bf16(false, a, false, bfr,
                                                    (short)0, acc, false, false);
    }
    const int col = dB + m;
    const float bc = b2[col];
#pragma unroll
    for (int v = 0; v < 8; ++v) {
      int token = tokenBase + v + hi;
      int b = token / NTOK, n = token % NTOK;
      size_t o = (size_t)n * (BDIM * DDIM) + (size_t)b * DDIM + col;
      out[o] = x[o] + acc[v] + bc;
    }
  }
}

// ---------------------------------------------------------------------------
extern "C" void kernel_launch(void* const* d_in, const int* in_sizes, int n_in,
                              void* d_out, int out_size, void* d_ws, size_t ws_size,
                              hipStream_t stream) {
  const float* x      = (const float*)d_in[0];
  const float* ln_w   = (const float*)d_in[1];
  const float* ln_b   = (const float*)d_in[2];
  const float* gamma  = (const float*)d_in[3];
  const float* gammax = (const float*)d_in[4];
  const float* w1     = (const float*)d_in[5];
  const float* b1     = (const float*)d_in[6];
  const float* w2     = (const float*)d_in[7];
  const float* b2     = (const float*)d_in[8];
  const float* dw3w   = (const float*)d_in[9];
  const float* dw3b   = (const float*)d_in[10];
  const float* dw5w   = (const float*)d_in[11];
  const float* dw5b   = (const float*)d_in[12];
  const float* dw7w   = (const float*)d_in[13];
  const float* dw7b   = (const float*)d_in[14];
  const float* projw  = (const float*)d_in[15];
  const float* projb  = (const float*)d_in[16];
  float* out = (float*)d_out;

  // workspace layout (all offsets 256B-aligned), ~42 MB total
  char* ws = (char*)d_ws;
  bf16_t* w1t  = (bf16_t*)(ws);                                  // 128 KB
  bf16_t* w2t  = (bf16_t*)(ws + 131072);                         // 128 KB
  float*  y1   = (float*) (ws + 262144);                         // 65600*64*4
  float*  mixf = (float*) (ws + 262144 + 16793600);              // 65536*64*4
  bf16_t* mixb = (bf16_t*)(ws + 262144 + 16793600 + 16777216);   // 65536*64*2

  k0_prep    <<<256,  256, 0, stream>>>(w1, w2, w1t, w2t);
  k1_ln_gemm <<<NTOTAL / 16, 128, 0, stream>>>(x, ln_w, ln_b, gamma, gammax,
                                               w1t, b1, y1);
  k2_conv    <<<(BDIM * 1024) / 16, 256, 0, stream>>>(y1, dw3w, dw3b, dw5w, dw5b,
                                                      dw7w, dw7b, mixf, mixb);
  k2b_proj   <<<(BDIM * 1024) / 16, 128, 0, stream>>>(mixb, projw, projb, mixf);
  k3_gelu_gemm<<<NTOTAL / 16, 256, 0, stream>>>(y1, mixf, w2t, b2, x, out);
}
